// RegionProposalNetwork_26405458936164
// MI455X (gfx1250) — compile-verified, matrix operand order checked by reading, not run on hardware
//
#include <hip/hip_runtime.h>
#include <math.h>

// ---------------- types ----------------
typedef __bf16 bf16;
typedef __attribute__((ext_vector_type(16))) __bf16 v16bf;
typedef __attribute__((ext_vector_type(8)))  __bf16 v8bf;
typedef __attribute__((ext_vector_type(8)))  float  v8f;

// ---------------- problem constants ----------------
#define HH    100
#define WW    152
#define NPOS  (HH*WW)          // 15200 spatial positions
#define NA    9
#define NBOX  (NPOS*NA)        // 136800 anchors
#define SORTN 262144           // 2^18 >= NBOX
#define NPRE  6000
#define NPOST 300
#define NBLK  ((NPRE+31)/32)   // 188 suppression words per row

// padded channel-last input: covers y-tiles 13x8 (+halo) and x-tiles 10x16 (+halo)
#define PADH2 106              // >= 13*8 + 1 + 1
#define PADW2 162              // >= 10*16 + 1 + 1
#define YTILES 13
#define XTILES 10
#define NSTEPS 144             // 16 K-blocks x 9 taps

// d_out layout (floats): rpn_loc [15200*36] | rpn_score [15200*18] | rois [300*4]
#define OUT_LOC_OFF   0
#define OUT_SCORE_OFF (NPOS*36)
#define OUT_ROIS_OFF  (NPOS*36 + NPOS*18)

__device__ __forceinline__ v8f zero8() {
  v8f z;
#pragma unroll
  for (int i = 0; i < 8; ++i) z[i] = 0.f;
  return z;
}

// Load a 16-bit WMMA fragment (A or B) from LDS, row-major [row][32] in K.
// CDNA5 16-bit A layout: lanes 0-15 hold K 0..7 / 16..23, lanes 16-31 hold
// K 8..15 / 24..31; caller selects the row with (lane & 15).
__device__ __forceinline__ v16bf ldfrag(const bf16* rowbase, int lane) {
  const int half = (lane >> 4) & 1;
  v8bf lo = *(const v8bf*)(rowbase + half * 8);        // K = half*8 .. +7
  v8bf hi = *(const v8bf*)(rowbase + 16 + half * 8);   // K = 16+half*8 .. +7
  v16bf r;
#pragma unroll
  for (int i = 0; i < 8; ++i) { r[i] = lo[i]; r[8 + i] = hi[i]; }
  return r;
}

// ---------------- prep: fp32 -> bf16 relayouts ----------------
// x (NCHW 1,512,100,152) -> channel-last padded xpad2[y][x][512]
__global__ void prep_x_kernel(const float* __restrict__ x, bf16* __restrict__ xpad2) {
  int t = blockIdx.x * blockDim.x + threadIdx.x;
  const int total = PADH2 * PADW2 * 512;
  if (t >= total) return;
  const int c = t & 511;
  const int rem = t >> 9;
  const int xx = rem % PADW2, yy = rem / PADW2;
  float v = 0.f;
  if (yy >= 1 && yy <= HH && xx >= 1 && xx <= WW)
    v = x[((size_t)c * HH + (yy - 1)) * WW + (xx - 1)];
  xpad2[t] = (bf16)v;
}

// conv1_w (OIHW 512,512,3,3) -> wc1t[tap s][oc][ic]  (K contiguous per B row)
__global__ void prep_w1_kernel(const float* __restrict__ w, bf16* __restrict__ wc1t) {
  int t = blockIdx.x * blockDim.x + threadIdx.x;
  const int total = 9 * 512 * 512;
  if (t >= total) return;
  const int c  = t & 511;
  const int oc = (t >> 9) & 511;
  const int s  = t >> 18;
  const int ky = s / 3, kx = s % 3;
  wc1t[t] = (bf16)w[(((size_t)oc * 512 + c) * 3 + ky) * 3 + kx];
}

// score_w (18,512) + loc_w (36,512) -> wht[n][ic], n<18 score, 18<=n<54 loc
__global__ void prep_wh_kernel(const float* __restrict__ sw, const float* __restrict__ lw,
                               bf16* __restrict__ wht) {
  int t = blockIdx.x * blockDim.x + threadIdx.x;
  const int total = 64 * 512;
  if (t >= total) return;
  const int c = t & 511;
  const int n = t >> 9;
  float v = 0.f;
  if (n < 18)      v = sw[(size_t)n * 512 + c];
  else if (n < 54) v = lw[(size_t)(n - 18) * 512 + c];
  wht[t] = (bf16)v;
}

// ---------------- conv1 3x3 as WMMA GEMM, double-buffered pipeline ----------------
// M-tile = 8 y-rows x 16 x-cols = 128 positions; N-tile = 128 out channels.
// Flattened 144-step loop (16 K-blocks x 9 taps): stage step n+1's B tile (and
// slab at K-block boundaries) into the other LDS buffer while step n's WMMAs
// consume the current one. One barrier per step.
__global__ __launch_bounds__(256, 2)
void conv1_wmma_kernel(const bf16* __restrict__ xpad2, const bf16* __restrict__ wc1t,
                       const float* __restrict__ bias, bf16* __restrict__ hout) {
  __shared__ __align__(16) bf16 slab[2][10 * 18 * 32];  // [buf][yy][xx][k]
  __shared__ __align__(16) bf16 Bs[2][128 * 32];        // [buf][n][k]
  const int tid  = threadIdx.x;
  const int lane = tid & 31;
  const int wave = tid >> 5;           // 8 waves
  const int waveM = wave & 3;          // 4 x 32-position slabs (2 subrows each)
  const int waveN = wave >> 2;         // 2 x 64-col slabs
  const int ytile = (int)blockIdx.x / XTILES;
  const int xtile = (int)blockIdx.x % XTILES;
  const int blockN = blockIdx.y * 128;

  v8f acc[2][4];
#pragma unroll
  for (int i = 0; i < 2; ++i)
#pragma unroll
    for (int j = 0; j < 4; ++j) acc[i][j] = zero8();

  // slab staging: thread t < 180 owns (yy,xx) pair, moves 32 contiguous channels
  const int syy = tid / 18, sxx = tid % 18;
  const bf16* slabsrc0 = xpad2 +
      ((size_t)(ytile * 8 + syy) * PADW2 + (xtile * 16 + sxx)) * 512;
  // B staging: thread owns (n, 16-k chunk)
  const int nB = tid >> 1, koB = (tid & 1) * 16;
  const bf16* wbase = wc1t + (size_t)(blockN + nB) * 512 + koB;

  // ---- prologue: stage step 0 ----
  if (tid < 180) {
    bf16* dst = &slab[0][tid * 32];
#pragma unroll
    for (int q = 0; q < 4; ++q)
      *(v8bf*)(dst + q * 8) = *(const v8bf*)(slabsrc0 + q * 8);
  }
  {
    const bf16* src = wbase;  // s=0, kc=0
    *(v8bf*)&Bs[0][nB * 32 + koB]     = *(const v8bf*)src;
    *(v8bf*)&Bs[0][nB * 32 + koB + 8] = *(const v8bf*)(src + 8);
  }

  for (int step = 0; step < NSTEPS; ++step) {
    const int kcIdx = step / 9;
    const int s = step - kcIdx * 9;
    __syncthreads();  // publish buffers staged for this step

    // ---- stage step+1 into the other buffers (overlaps with WMMA below) ----
    if (step + 1 < NSTEPS) {
      const int n1 = step + 1;
      const int kcIdx1 = n1 / 9;
      const int s1 = n1 - kcIdx1 * 9;
      {
        const bf16* src = wbase + ((size_t)s1 * 512) * 512 + kcIdx1 * 32;
        bf16* dst = &Bs[n1 & 1][nB * 32 + koB];
        *(v8bf*)dst       = *(const v8bf*)src;
        *(v8bf*)(dst + 8) = *(const v8bf*)(src + 8);
      }
      if (s1 == 0 && tid < 180) {
        const bf16* src = slabsrc0 + kcIdx1 * 32;
        bf16* dst = &slab[kcIdx1 & 1][tid * 32];
#pragma unroll
        for (int q = 0; q < 4; ++q)
          *(v8bf*)(dst + q * 8) = *(const v8bf*)(src + q * 8);
      }
      if (step + 2 < NSTEPS) {
        const int n2 = step + 2;
        const int kcIdx2 = n2 / 9;
        const int s2 = n2 - kcIdx2 * 9;
        __builtin_prefetch(wbase + ((size_t)s2 * 512) * 512 + kcIdx2 * 32, 0, 1);
      }
    }

    // ---- consume current buffers: 2 A-frags x 4 B-frags -> 8 WMMA ----
    const bf16* sl = slab[kcIdx & 1];
    const bf16* Bp = Bs[step & 1];
    const int dy = s / 3, dx = s % 3;   // tap offsets (0..2 == -1..1 + halo)
    v16bf af[2], bfr[4];
#pragma unroll
    for (int i = 0; i < 2; ++i) {
      const int subrow = waveM * 2 + i;
      const int xi = lane & 15;
      af[i] = ldfrag(sl + ((subrow + dy) * 18 + (xi + dx)) * 32, lane);
    }
#pragma unroll
    for (int j = 0; j < 4; ++j)
      bfr[j] = ldfrag(Bp + (waveN * 64 + j * 16 + (lane & 15)) * 32, lane);
#pragma unroll
    for (int i = 0; i < 2; ++i)
#pragma unroll
      for (int j = 0; j < 4; ++j)
        acc[i][j] = __builtin_amdgcn_wmma_f32_16x16x32_bf16(
            false, af[i], false, bfr[j], (short)0, acc[i][j], false, false);
  }

  // epilogue: bias + ReLU -> bf16 h[pos][512] (K-contiguous for head GEMM)
#pragma unroll
  for (int i = 0; i < 2; ++i) {
    const int subrow = waveM * 2 + i;
    const int y = ytile * 8 + subrow;
#pragma unroll
    for (int j = 0; j < 4; ++j) {
      const int col = blockN + waveN * 64 + j * 16 + (lane & 15);
      const float b = bias[col];
#pragma unroll
      for (int r = 0; r < 8; ++r) {
        const int xi = ((lane >> 4) << 3) + r;
        const int x = xtile * 16 + xi;
        if (y < HH && x < WW) {
          float v = acc[i][j][r] + b;
          v = v > 0.f ? v : 0.f;
          hout[((size_t)y * WW + x) * 512 + col] = (bf16)v;
        }
      }
    }
  }
}

// ---------------- fused score+loc 1x1 head: [15200 x 512] x [512 x 64] ----------------
__global__ __launch_bounds__(256, 2)
void head_wmma_kernel(const bf16* __restrict__ h, const bf16* __restrict__ wht,
                      const float* __restrict__ score_b, const float* __restrict__ loc_b,
                      float* __restrict__ out_loc, float* __restrict__ out_score) {
  __shared__ __align__(16) bf16 As[128 * 32];   // [m][k]
  __shared__ __align__(16) bf16 Bs[64 * 32];    // [n][k]
  const int tid  = threadIdx.x;
  const int lane = tid & 31;
  const int wave = tid >> 5;             // 8 waves x 16 rows each
  const int blockM = blockIdx.x * 128;

  v8f acc[4];
#pragma unroll
  for (int j = 0; j < 4; ++j) acc[j] = zero8();

  const int ma = tid >> 1, ka = (tid & 1) * 16;   // A: 128 m x 32 k
  const int nB = tid >> 2, koB = (tid & 3) * 8;   // B: 64 n x 32 k

  for (int kc = 0; kc < 512; kc += 32) {
    {
      int m = blockM + ma;
      m = m < NPOS ? m : NPOS - 1;
      const bf16* src = h + (size_t)m * 512 + kc + ka;
      *(v8bf*)&As[ma * 32 + ka]     = *(const v8bf*)src;
      *(v8bf*)&As[ma * 32 + ka + 8] = *(const v8bf*)(src + 8);
    }
    {
      const bf16* src = wht + (size_t)nB * 512 + kc + koB;
      *(v8bf*)&Bs[nB * 32 + koB] = *(const v8bf*)src;
    }
    __syncthreads();
    v16bf af = ldfrag(&As[(wave * 16 + (lane & 15)) * 32], lane);
#pragma unroll
    for (int j = 0; j < 4; ++j) {
      v16bf bfr = ldfrag(&Bs[(j * 16 + (lane & 15)) * 32], lane);
      acc[j] = __builtin_amdgcn_wmma_f32_16x16x32_bf16(
          false, af, false, bfr, (short)0, acc[j], false, false);
    }
    __syncthreads();
  }

  const int rbase = blockM + wave * 16 + ((lane >> 4) << 3);
#pragma unroll
  for (int j = 0; j < 4; ++j) {
    const int col = j * 16 + (lane & 15);
#pragma unroll
    for (int r = 0; r < 8; ++r) {
      const int row = rbase + r;
      if (row < NPOS) {
        const float v = acc[j][r];
        if (col < 18)       out_score[(size_t)row * 18 + col] = v + score_b[col];
        else if (col < 54)  out_loc[(size_t)row * 36 + (col - 18)] = v + loc_b[col - 18];
      }
    }
  }
}

// ---------------- anchor decode + clip + validity + sort key ----------------
__global__ void decode_kernel(const float* __restrict__ out_loc,
                              const float* __restrict__ out_score,
                              const int* __restrict__ img_h_p, const int* __restrict__ img_w_p,
                              float* __restrict__ roi,
                              unsigned long long* __restrict__ key) {
  int t = blockIdx.x * blockDim.x + threadIdx.x;
  if (t >= SORTN) return;
  if (t >= NBOX) { key[t] = 0ull; return; }  // pad: sorts below everything
  const int p = t / NA, a = t % NA;
  const int y = p / WW, x = p % WW;
  const int ri = a / 3, si = a % 3;
  const float ratios[3] = {0.5f, 1.f, 2.f};
  const float scales[3] = {8.f, 16.f, 32.f};
  const float hA = 16.f * scales[si] * sqrtf(ratios[ri]);
  const float wA = 16.f * scales[si] * sqrtf(1.f / ratios[ri]);
  const float cy = y * 16.f + 8.f, cx = x * 16.f + 8.f;
  const float dy  = out_loc[(size_t)p * 36 + a * 4 + 0];
  const float dxv = out_loc[(size_t)p * 36 + a * 4 + 1];
  const float dh  = out_loc[(size_t)p * 36 + a * 4 + 2];
  const float dw  = out_loc[(size_t)p * 36 + a * 4 + 3];
  const float s0 = out_score[(size_t)p * 18 + a * 2 + 0];
  const float s1 = out_score[(size_t)p * 18 + a * 2 + 1];
  const float fg = 1.f / (1.f + __expf(s0 - s1));   // softmax fg prob
  const float cy2 = dy * hA + cy, cx2 = dxv * wA + cx;
  const float h2 = __expf(dh) * hA, w2 = __expf(dw) * wA;
  const float imh = (float)img_h_p[0], imw = (float)img_w_p[0];
  const float y1  = fminf(fmaxf(cy2 - 0.5f * h2, 0.f), imh);
  const float x1  = fminf(fmaxf(cx2 - 0.5f * w2, 0.f), imw);
  const float y2c = fminf(fmaxf(cy2 + 0.5f * h2, 0.f), imh);
  const float x2c = fminf(fmaxf(cx2 + 0.5f * w2, 0.f), imw);
  roi[(size_t)t * 4 + 0] = y1;  roi[(size_t)t * 4 + 1] = x1;
  roi[(size_t)t * 4 + 2] = y2c; roi[(size_t)t * 4 + 3] = x2c;
  const bool valid = (y2c - y1 >= 16.f) && (x2c - x1 >= 16.f);
  const float sc = valid ? fg : -__builtin_inff();
  unsigned u = __float_as_uint(sc);
  unsigned k32 = (u & 0x80000000u) ? ~u : (u | 0x80000000u);  // monotone key
  // descending sort; tie-break = ascending original index (matches stable top_k)
  key[t] = ((unsigned long long)k32 << 32) | (unsigned)(0xFFFFFFFFu - (unsigned)t);
}

// ---------------- bitonic sort (descending), one compare-exchange pass ----------------
__global__ void bitonic_step_kernel(unsigned long long* __restrict__ key, int j, int k) {
  int i = blockIdx.x * blockDim.x + threadIdx.x;
  int l = i ^ j;
  if (l > i) {
    unsigned long long a = key[i], b = key[l];
    const bool upblk = ((i & k) == 0);
    const bool swp = upblk ? (a < b) : (a > b);  // descending overall
    if (swp) { key[i] = b; key[l] = a; }
  }
}

// ---------------- gather top-6000 boxes in score order ----------------
__global__ void gather_kernel(const unsigned long long* __restrict__ key,
                              const float* __restrict__ roi,
                              float* __restrict__ roi_s, int* __restrict__ valid) {
  int i = blockIdx.x * blockDim.x + threadIdx.x;
  if (i >= NPRE) return;
  const unsigned long long kk = key[i];
  unsigned idx = 0xFFFFFFFFu - (unsigned)(kk & 0xFFFFFFFFu);
  const unsigned k32 = (unsigned)(kk >> 32);
  if (idx >= NBOX) idx = 0;  // pads can't reach the top 6000; safety only
#pragma unroll
  for (int c = 0; c < 4; ++c) roi_s[(size_t)i * 4 + c] = roi[(size_t)idx * 4 + c];
  valid[i] = (k32 & 0x80000000u) ? 1 : 0;  // score > -inf
}

// ---------------- NMS suppression bitmask ----------------
__global__ void nms_mask_kernel(const float* __restrict__ roi_s, unsigned* __restrict__ mask) {
  int t = blockIdx.x * blockDim.x + threadIdx.x;
  if (t >= NPRE * NBLK) return;
  const int i = t / NBLK, blk = t % NBLK;
  const float y1 = roi_s[(size_t)i * 4 + 0], x1 = roi_s[(size_t)i * 4 + 1];
  const float y2 = roi_s[(size_t)i * 4 + 2], x2 = roi_s[(size_t)i * 4 + 3];
  const float ai = (y2 - y1) * (x2 - x1);
  unsigned m = 0u;
  const int jbase = blk * 32;
#pragma unroll 4
  for (int jj = 0; jj < 32; ++jj) {
    const int j = jbase + jj;
    if (j < NPRE && j > i) {
      const float by1 = roi_s[(size_t)j * 4 + 0], bx1 = roi_s[(size_t)j * 4 + 1];
      const float by2 = roi_s[(size_t)j * 4 + 2], bx2 = roi_s[(size_t)j * 4 + 3];
      float ih = fminf(y2, by2) - fmaxf(y1, by1); ih = fmaxf(ih, 0.f);
      float iw = fminf(x2, bx2) - fmaxf(x1, bx1); iw = fmaxf(iw, 0.f);
      const float inter = ih * iw;
      const float aj = (by2 - by1) * (bx2 - bx1);
      const float iou = inter / (ai + aj - inter + 1e-9f);
      if (iou > 0.7f) m |= (1u << jj);
    }
  }
  mask[(size_t)i * NBLK + blk] = m;
}

// ---------------- sequential greedy reduce (single block) ----------------
// Reference semantics: an unsuppressed box suppresses later overlaps even if
// its own score is -inf; validity gates only the final keep flag.
__global__ void nms_reduce_kernel(const unsigned* __restrict__ mask,
                                  const int* __restrict__ valid, int* __restrict__ keep) {
  __shared__ unsigned remv[NBLK];
  __shared__ int cur;
  const int t = threadIdx.x;
  if (t < NBLK) remv[t] = 0u;
  __syncthreads();
  for (int i = 0; i < NPRE; ++i) {
    if (t == 0) cur = ((remv[i >> 5] >> (i & 31)) & 1u) ? 0 : 1;
    __syncthreads();
    if (cur && t < NBLK) remv[t] |= mask[(size_t)i * NBLK + t];
    if (t == 0) keep[i] = cur && valid[i];
    __syncthreads();
  }
}

// ---------------- take first 300 kept (== top-300 by score) ----------------
__global__ void finalize_kernel(const float* __restrict__ roi_s,
                                const int* __restrict__ keep, float* __restrict__ rois) {
  if (threadIdx.x != 0 || blockIdx.x != 0) return;
  int cnt = 0;
  for (int i = 0; i < NPRE && cnt < NPOST; ++i) {
    if (keep[i]) {
      rois[cnt * 4 + 0] = roi_s[(size_t)i * 4 + 0];
      rois[cnt * 4 + 1] = roi_s[(size_t)i * 4 + 1];
      rois[cnt * 4 + 2] = roi_s[(size_t)i * 4 + 2];
      rois[cnt * 4 + 3] = roi_s[(size_t)i * 4 + 3];
      ++cnt;
    }
  }
  for (; cnt < NPOST; ++cnt) {
    rois[cnt * 4 + 0] = 0.f; rois[cnt * 4 + 1] = 0.f;
    rois[cnt * 4 + 2] = 0.f; rois[cnt * 4 + 3] = 0.f;
  }
}

// ---------------- host launcher ----------------
extern "C" void kernel_launch(void* const* d_in, const int* in_sizes, int n_in,
                              void* d_out, int out_size, void* d_ws, size_t ws_size,
                              hipStream_t stream) {
  (void)in_sizes; (void)n_in; (void)out_size; (void)ws_size;
  const float* x       = (const float*)d_in[0];
  const float* conv1_w = (const float*)d_in[1];
  const float* conv1_b = (const float*)d_in[2];
  const float* score_w = (const float*)d_in[3];
  const float* score_b = (const float*)d_in[4];
  const float* loc_w   = (const float*)d_in[5];
  const float* loc_b   = (const float*)d_in[6];
  const int*   img_h   = (const int*)d_in[7];
  const int*   img_w   = (const int*)d_in[8];

  float* out       = (float*)d_out;
  float* out_loc   = out + OUT_LOC_OFF;
  float* out_score = out + OUT_SCORE_OFF;
  float* out_rois  = out + OUT_ROIS_OFF;

  // workspace partition (256B aligned)
  char* base = (char*)d_ws;
  size_t off = 0;
  auto carve = [&](size_t bytes) -> char* {
    char* p = base + off;
    off = (off + bytes + 255) & ~(size_t)255;
    return p;
  };
  bf16* xpad2 = (bf16*)carve((size_t)PADH2 * PADW2 * 512 * sizeof(bf16));   // 17.6 MB
  bf16* wc1t  = (bf16*)carve((size_t)9 * 512 * 512 * sizeof(bf16));         //  4.7 MB
  bf16* wht   = (bf16*)carve((size_t)64 * 512 * sizeof(bf16));
  bf16* hbuf  = (bf16*)carve((size_t)NPOS * 512 * sizeof(bf16));            // 15.6 MB
  float* roi  = (float*)carve((size_t)NBOX * 4 * sizeof(float));            //  2.2 MB
  unsigned long long* key = (unsigned long long*)carve((size_t)SORTN * 8);  //  2.1 MB
  float* roi_s = (float*)carve((size_t)NPRE * 4 * sizeof(float));
  int* valid   = (int*)carve((size_t)NPRE * sizeof(int));
  unsigned* mask = (unsigned*)carve((size_t)NPRE * NBLK * sizeof(unsigned)); // 4.5 MB
  int* keep    = (int*)carve((size_t)NPRE * sizeof(int));

  // 1) precision/layout prep
  {
    const int t1 = PADH2 * PADW2 * 512;
    prep_x_kernel<<<(t1 + 255) / 256, 256, 0, stream>>>(x, xpad2);
    const int t2 = 9 * 512 * 512;
    prep_w1_kernel<<<(t2 + 255) / 256, 256, 0, stream>>>(conv1_w, wc1t);
    const int t3 = 64 * 512;
    prep_wh_kernel<<<(t3 + 255) / 256, 256, 0, stream>>>(score_w, loc_w, wht);
  }

  // 2) conv1 (3x3, 512->512) + bias + ReLU via bf16 WMMA (double-buffered pipeline)
  {
    dim3 grid(YTILES * XTILES, 512 / 128);  // 130 x 4
    conv1_wmma_kernel<<<grid, 256, 0, stream>>>(xpad2, wc1t, conv1_b, hbuf);
  }

  // 3) fused 1x1 score+loc heads via bf16 WMMA (writes rpn_loc / rpn_score)
  head_wmma_kernel<<<(NPOS + 127) / 128, 256, 0, stream>>>(
      hbuf, wht, score_b, loc_b, out_loc, out_score);

  // 4) anchor decode + clip + validity + sort keys
  decode_kernel<<<SORTN / 256, 256, 0, stream>>>(out_loc, out_score, img_h, img_w, roi, key);

  // 5) full descending bitonic sort of 2^18 keys (deterministic launch sequence)
  for (int k = 2; k <= SORTN; k <<= 1)
    for (int j = k >> 1; j > 0; j >>= 1)
      bitonic_step_kernel<<<SORTN / 256, 256, 0, stream>>>(key, j, k);

  // 6) gather top-6000, NMS bitmask, sequential reduce, emit 300 rois
  gather_kernel<<<(NPRE + 255) / 256, 256, 0, stream>>>(key, roi, roi_s, valid);
  nms_mask_kernel<<<(NPRE * NBLK + 255) / 256, 256, 0, stream>>>(roi_s, mask);
  nms_reduce_kernel<<<1, 224, 0, stream>>>(mask, valid, keep);
  finalize_kernel<<<1, 32, 0, stream>>>(roi_s, keep, out_rois);
}
